// LSTMText_46291157516875
// MI455X (gfx1250) — matrix operand order repaired and checked by reading, not compiled, not run on hardware
//
#include <hip/hip_runtime.h>
#include <math.h>

// Problem dims (fixed by the reference)
#define Bb 64
#define Ss 512
#define Ii 1024
#define Hh 1024
#define KK 2048    // I + H (fused input+recurrent GEMM K)
#define NG 4096    // 4*H gate columns

#define NBLK 64    // persistent workgroups (>= co-resident requirement: 64 WGPs)
#define LROW 2056  // LDS row pitch in f16 (2048 + 8 pad -> uniform 2-way banks)
#define WLDS_BYTES ((size_t)64 * LROW * 2)          // 263,168 B weight slice
#define GSH_BYTES  ((size_t)4 * 64 * 16 * 4)        // 16 KB gate exchange
#define SMEM_BYTES (WLDS_BYTES + GSH_BYTES)         // 279,552 B <= 320 KB

typedef __attribute__((ext_vector_type(16))) _Float16 v16h;
typedef __attribute__((ext_vector_type(8)))  _Float16 v8h;
typedef __attribute__((ext_vector_type(4)))  _Float16 v4h;
typedef __attribute__((ext_vector_type(8)))  float    v8f;
typedef __attribute__((ext_vector_type(4)))  float    v4f;

// ---------------- prep kernels (run once per call) ----------------

__global__ void k_zero_state(_Float16* __restrict__ h0, _Float16* __restrict__ h1,
                             float* __restrict__ c0, unsigned* __restrict__ cnt) {
    int i = blockIdx.x * blockDim.x + threadIdx.x;
    if (i < Bb * Hh) { h0[i] = (_Float16)0.f; h1[i] = (_Float16)0.f; c0[i] = 0.f; }
    if (i == 0) *cnt = 0u;   // reset persistent-kernel step barrier each call
}

// X f32 -> f16, same [B,S,I] layout. 4 elements/thread.
__global__ void k_cvt_x(const float* __restrict__ X, _Float16* __restrict__ Xh) {
    size_t i = ((size_t)blockIdx.x * blockDim.x + threadIdx.x) * 4;
    v4f x = *(const v4f*)(X + i);
    v4h y;
    y[0] = (_Float16)x[0]; y[1] = (_Float16)x[1];
    y[2] = (_Float16)x[2]; y[3] = (_Float16)x[3];
    *(v4h*)(Xh + i) = y;
}

// Build transposed fused weight: Wt[n][k], n in [0,4096) gate-major,
// k in [0,2048): k<1024 -> Wx_gate[k][n%1024], else Wh_gate[k-1024][n%1024].
__global__ void k_build_wt(const float* __restrict__ wx0, const float* __restrict__ wx1,
                           const float* __restrict__ wx2, const float* __restrict__ wx3,
                           const float* __restrict__ wh0, const float* __restrict__ wh1,
                           const float* __restrict__ wh2, const float* __restrict__ wh3,
                           _Float16* __restrict__ Wt) {
    size_t tid = (size_t)blockIdx.x * blockDim.x + threadIdx.x;   // 4096*2048 threads
    int k    = (int)(tid & (KK - 1));
    int n    = (int)(tid >> 11);
    int gate = n >> 10;
    int nc   = n & (Hh - 1);
    const float* wx = (gate == 0) ? wx0 : (gate == 1) ? wx1 : (gate == 2) ? wx2 : wx3;
    const float* wh = (gate == 0) ? wh0 : (gate == 1) ? wh1 : (gate == 2) ? wh2 : wh3;
    float v = (k < Ii) ? wx[(size_t)k * Hh + nc]
                       : wh[(size_t)(k - Ii) * Hh + nc];
    Wt[tid] = (_Float16)v;
}

// H_f = result[:, S-1, :]
__global__ void k_final_h(const float* __restrict__ out, float* __restrict__ Hf) {
    int i = blockIdx.x * blockDim.x + threadIdx.x;   // over B*H
    int m = i >> 10, h = i & (Hh - 1);
    Hf[i] = out[((size_t)m * Ss + (Ss - 1)) * Hh + h];
}

// ---------------- persistent LSTM kernel ----------------

__device__ __forceinline__ float sigm(float x) { return 1.f / (1.f + __expf(-x)); }
__device__ __forceinline__ float tanh_fast(float x) { return 2.f / (1.f + __expf(-2.f * x)) - 1.f; }

// Build A 16x32 f16 fragment from a row-major f16 row pointer (row = lane%16).
// Layout: e<8 -> K = hv*8+e ; e>=8 -> K = 16+hv*8+(e-8).
__device__ __forceinline__ v16h load_a(const _Float16* p, int hv) {
    v8h lo = *(const v8h*)(p + hv * 8);
    v8h hi = *(const v8h*)(p + 16 + hv * 8);
    return __builtin_shufflevector(lo, hi, 0,1,2,3,4,5,6,7,8,9,10,11,12,13,14,15);
}

// Grid: 64 persistent blocks x 256 threads (8 waves). Block b owns h-columns
// [b*16, b*16+16) of ALL 4 gates; its weight slice (64 rows x 2048 K f16 =
// 256 KB) is pinned in LDS for the whole sequence. Wave w = gate (w>>1) and
// m-half (w&1); each wave computes 2 m-tiles x 1 n-tile per step (K = 2048).
// A device-wide release/acquire counter barrier separates timesteps.
__global__ __launch_bounds__(256) void k_lstm_persist(
    const _Float16* __restrict__ Xh,   // [B,S,I] f16
    const _Float16* __restrict__ Wt,   // [4096][2048] f16 (N-major, K contig)
    const float* __restrict__ bi, const float* __restrict__ bff,
    const float* __restrict__ bo, const float* __restrict__ bc,
    _Float16* __restrict__ Hb0,        // f16 H state buffer 0 [64][1024]
    _Float16* __restrict__ Hb1,        // f16 H state buffer 1 [64][1024]
    float* __restrict__ Cs,            // C state (in-place)   [64][1024] f32
    float* __restrict__ out,           // [B,S,H] f32
    unsigned* __restrict__ cnt)        // step barrier counter (pre-zeroed)
{
    extern __shared__ char smem[];
    _Float16* wlds = (_Float16*)smem;                       // [64][LROW]
    float (*gsh)[64][16] = (float (*)[64][16])(smem + WLDS_BYTES);  // [4][64][16]

    const int tid  = threadIdx.x;
    const int lane = tid & 31;
    const int w    = tid >> 5;          // wave id
    const int gate = w >> 1;
    const int mh   = w & 1;             // m-half: rows [mh*32, mh*32+32)
    const int hv   = lane >> 4;         // lane half (K-half selector)
    const int l16  = lane & 15;
    const int hg   = blockIdx.x;        // 16-wide h-column group (0..63)

    // ---- pin this block's weight slice in LDS (once) ----
    // LDS row r = gate*16 + l  <->  global row n = gate*1024 + hg*16 + l
    for (int idx = tid; idx < 64 * 128; idx += 256) {
        int r = idx >> 7, seg = idx & 127;
        int g = r >> 4, l = r & 15;
        v16h v = *(const v16h*)(Wt + (size_t)(g * Hh + hg * 16 + l) * KK + seg * 16);
        *(v16h*)(wlds + (size_t)r * LROW + seg * 16) = v;
    }
    __syncthreads();

    // per-wave constant offsets
    const size_t xoff0 = (size_t)(mh * 32 + l16) * Ss * Ii;        // m-tile 0 row
    const size_t xoff1 = (size_t)(mh * 32 + 16 + l16) * Ss * Ii;   // m-tile 1 row
    const size_t hoff0 = (size_t)(mh * 32 + l16) * Hh;
    const size_t hoff1 = (size_t)(mh * 32 + 16 + l16) * Hh;
    const _Float16* brow = wlds + (size_t)(gate * 16 + l16) * LROW + hv * 8;

    for (int t = 0; t < Ss; ++t) {
        const _Float16* Hr = (t & 1) ? Hb1 : Hb0;
        _Float16*       Hw = (t & 1) ? Hb0 : Hb1;
        const _Float16* a0p = Xh + xoff0 + (size_t)t * Ii;
        const _Float16* a1p = Xh + xoff1 + (size_t)t * Ii;
        const _Float16* h0p = Hr + hoff0;
        const _Float16* h1p = Hr + hoff1;

        v8f acc0 = {0.f,0.f,0.f,0.f,0.f,0.f,0.f,0.f};
        v8f acc1 = acc0;

        // K = 0..1023 : input projection (A from f16 X, B from LDS)
        for (int k0 = 0; k0 < Ii; k0 += 32) {
            v16h A0 = load_a(a0p + k0, hv);
            v16h A1 = load_a(a1p + k0, hv);
            v16h B  = __builtin_shufflevector(*(const v8h*)(brow + k0), *(const v8h*)(brow + k0 + 16), 0,1,2,3,4,5,6,7,8,9,10,11,12,13,14,15);
            acc0 = __builtin_amdgcn_wmma_f32_16x16x32_f16(false, A0, false, B, (short)0, acc0, false, false);
            acc1 = __builtin_amdgcn_wmma_f32_16x16x32_f16(false, A1, false, B, (short)0, acc1, false, false);
        }
        // K = 1024..2047 : recurrent part (A from f16 H state, B from LDS)
        for (int k0 = 0; k0 < Hh; k0 += 32) {
            v16h A0 = load_a(h0p + k0, hv);
            v16h A1 = load_a(h1p + k0, hv);
            v16h B  = __builtin_shufflevector(*(const v8h*)(brow + Ii + k0), *(const v8h*)(brow + Ii + k0 + 16), 0,1,2,3,4,5,6,7,8,9,10,11,12,13,14,15);
            acc0 = __builtin_amdgcn_wmma_f32_16x16x32_f16(false, A0, false, B, (short)0, acc0, false, false);
            acc1 = __builtin_amdgcn_wmma_f32_16x16x32_f16(false, A1, false, B, (short)0, acc1, false, false);
        }

        // D layout: element r -> row = r + 8*hv, col = l16 (per 16x16 tile)
        #pragma unroll
        for (int r = 0; r < 8; ++r) {
            gsh[gate][mh * 32 + r + hv * 8     ][l16] = acc0[r];
            gsh[gate][mh * 32 + r + hv * 8 + 16][l16] = acc1[r];
        }
        __syncthreads();

        // fused LSTM gate math: 64x16 elements over 256 threads
        for (int e = tid; e < 64 * 16; e += 256) {
            int row = e >> 4, col = e & 15;
            int m = row;
            int h = hg * 16 + col;
            float gi = gsh[0][row][col] + bi[h];
            float gf = gsh[1][row][col] + bff[h];
            float go = gsh[2][row][col] + bo[h];
            float gc = gsh[3][row][col] + bc[h];
            float it = sigm(gi);
            float ft = sigm(gf);
            float ot = sigm(go);
            float ct = tanh_fast(gc);
            size_t sidx = (size_t)m * Hh + h;
            float cn = ft * Cs[sidx] + it * ct;
            float hn = ot * tanh_fast(cn);
            Cs[sidx] = cn;
            Hw[sidx] = (_Float16)hn;
            out[((size_t)m * Ss + t) * Hh + h] = hn;
        }

        // ---- device-wide step barrier (release / acquire) ----
        __threadfence();      // my H/C/out stores visible at agent scope
        __syncthreads();      // whole block done with step t (incl. gsh reads)
        if (tid == 0) {
            __hip_atomic_fetch_add(cnt, 1u, __ATOMIC_RELEASE, __HIP_MEMORY_SCOPE_AGENT);
            unsigned target = (unsigned)NBLK * (unsigned)(t + 1);
            while (__hip_atomic_load(cnt, __ATOMIC_ACQUIRE, __HIP_MEMORY_SCOPE_AGENT) < target) {
                __builtin_amdgcn_s_sleep(1);
            }
        }
        __syncthreads();
        __threadfence();      // acquire ordering for this thread's next-step loads
    }
}

// ---------------- host side ----------------

extern "C" void kernel_launch(void* const* d_in, const int* in_sizes, int n_in,
                              void* d_out, int out_size, void* d_ws, size_t ws_size,
                              hipStream_t stream) {
    const float* X   = (const float*)d_in[0];
    const float* Wxi = (const float*)d_in[1];
    const float* Whi = (const float*)d_in[2];
    const float* bi  = (const float*)d_in[3];
    const float* Wxf = (const float*)d_in[4];
    const float* Whf = (const float*)d_in[5];
    const float* bfv = (const float*)d_in[6];
    const float* Wxo = (const float*)d_in[7];
    const float* Who = (const float*)d_in[8];
    const float* bo  = (const float*)d_in[9];
    const float* Wxc = (const float*)d_in[10];
    const float* Whc = (const float*)d_in[11];
    const float* bc  = (const float*)d_in[12];

    float* out = (float*)d_out;
    // Output tuple layout: [result B*S*H][H_f B*H][C_f B*H].
    float* Hf = out + (size_t)Bb * Ss * Hh;   // written by k_final_h at the end
    float* C0 = Hf + (size_t)Bb * Hh;         // live f32 C state == C_f output

    // Workspace: [Wt f16 16MB][X f16 64MB][H f16 x2 256KB][barrier counter]
    _Float16* Wt  = (_Float16*)d_ws;
    _Float16* Xh  = (_Float16*)((char*)d_ws + (size_t)NG * KK * sizeof(_Float16));
    _Float16* Hb0 = (_Float16*)((char*)Xh + (size_t)Bb * Ss * Ii * sizeof(_Float16));
    _Float16* Hb1 = Hb0 + (size_t)Bb * Hh;
    unsigned* cnt = (unsigned*)(Hb1 + (size_t)Bb * Hh);

    k_zero_state<<<(Bb * Hh + 255) / 256, 256, 0, stream>>>(Hb0, Hb1, C0, cnt);
    k_cvt_x<<<(int)(((size_t)Bb * Ss * Ii / 4) / 256), 256, 0, stream>>>(X, Xh);
    k_build_wt<<<(int)(((size_t)NG * KK) / 256), 256, 0, stream>>>(
        Wxi, Wxf, Wxo, Wxc, Whi, Whf, Who, Whc, Wt);

    // One persistent launch runs all 512 timesteps; weights live in LDS.
    k_lstm_persist<<<NBLK, 256, SMEM_BYTES, stream>>>(
        Xh, Wt, bi, bfv, bo, bc, Hb0, Hb1, C0, out, cnt);

    // H_f = result[:, S-1, :]
    k_final_h<<<(Bb * Hh) / 256, 256, 0, stream>>>(out, Hf);
}
